// MultiHeadAttention_48584670052354
// MI455X (gfx1250) — matrix operand order repaired and checked by reading, not compiled, not run on hardware
//
#include <hip/hip_runtime.h>

typedef __attribute__((ext_vector_type(16))) __bf16 bf16x16;
typedef __attribute__((ext_vector_type(8)))  __bf16 bf16x8;
typedef __attribute__((ext_vector_type(8)))  float  f32x8;
typedef __attribute__((ext_vector_type(4)))  int    v4i;

#define B_   4
#define L_   1024
#define H_   16
#define D_   64
#define HID_ 1024

#ifndef __has_builtin
#define __has_builtin(x) 0
#endif
#if __has_builtin(__builtin_amdgcn_global_load_async_to_lds_b128)
#define HAVE_ASYNC_B128 1
#else
#define HAVE_ASYNC_B128 0
#endif

// ---------------------------------------------------------------------------
// D = A(16x32 bf16) * B(32x16 bf16) + C(16x16 f32)
__device__ __forceinline__ f32x8 wmma_bf16(bf16x16 a, bf16x16 b, f32x8 c) {
  return __builtin_amdgcn_wmma_f32_16x16x32_bf16(false, a, false, b, (short)0, c,
                                                 false, false);
}

__device__ __forceinline__ bf16x16 frag_pair(const __bf16* p0, const __bf16* p1) {
  union { bf16x8 h[2]; bf16x16 f; } u;
  u.h[0] = *(const bf16x8*)p0;
  u.h[1] = *(const bf16x8*)p1;
  return u.f;
}
// B fragment: per-lane 16 contiguous K values (lane = column, halves split K).
__device__ __forceinline__ bf16x16 frag_b(const __bf16* p) { return frag_pair(p, p + 8); }
// A fragment: per-lane row pointer; hi8 = 8 for lanes 16..31 (K 16-31).
__device__ __forceinline__ bf16x16 frag_a(const __bf16* row, int hi8) {
  return frag_pair(row + 2 * hi8, row + 2 * hi8 + 8);
}

// 16-byte global -> LDS staging; async (ASYNCcnt) when the builtin exists.
__device__ __forceinline__ void stage16(const __bf16* g, __bf16* l) {
#if HAVE_ASYNC_B128
  __builtin_amdgcn_global_load_async_to_lds_b128((v4i*)g, (v4i*)l, 0, 0);
#else
  *(bf16x8*)l = *(const bf16x8*)g;
#endif
}
template <int N>
__device__ __forceinline__ void stage_wait_n() {
#if HAVE_ASYNC_B128
#if __has_builtin(__builtin_amdgcn_s_wait_asynccnt)
  __builtin_amdgcn_s_wait_asynccnt(N);
#else
  asm volatile("s_wait_asynccnt %0" ::"i"(N) : "memory");
#endif
#endif
}

// ---------------------------------------------------------------------------
__global__ __launch_bounds__(256) void cvt_bf16_kernel(const float* __restrict__ in,
                                                       __bf16* __restrict__ out, int n) {
  int i = blockIdx.x * 256 + threadIdx.x;
  if (i < n) out[i] = (__bf16)in[i];
}

// Convert the three input activation tensors to bf16 (contiguous outputs).
__global__ __launch_bounds__(256) void cvt_x_kernel(const float* __restrict__ x0,
                                                    const float* __restrict__ x1,
                                                    const float* __restrict__ x2,
                                                    __bf16* __restrict__ out) {
  const float* x = (blockIdx.z == 0) ? x0 : ((blockIdx.z == 1) ? x1 : x2);
  __bf16* o = out + (size_t)blockIdx.z * ((size_t)B_ * L_ * HID_);
  size_t i = ((size_t)blockIdx.x * 256 + threadIdx.x) * 8;
  float4 a = *(const float4*)(x + i);
  float4 b = *(const float4*)(x + i + 4);
  bf16x8 r = {(__bf16)a.x, (__bf16)a.y, (__bf16)a.z, (__bf16)a.w,
              (__bf16)b.x, (__bf16)b.y, (__bf16)b.z, (__bf16)b.w};
  *(bf16x8*)(o + i) = r;
}

// Convert + transpose a weight matrix: out[n][k] (bf16) = in[k][n] (f32).
__global__ __launch_bounds__(256) void cvt_wT_kernel(
    const float* __restrict__ w0, const float* __restrict__ w1,
    const float* __restrict__ w2, const float* __restrict__ w3,
    __bf16* __restrict__ o0, __bf16* __restrict__ o1,
    __bf16* __restrict__ o2, __bf16* __restrict__ o3) {
  __shared__ float tile[32][33];
  const float* w;
  __bf16* o;
  switch (blockIdx.z) {
    case 0:  w = w0; o = o0; break;
    case 1:  w = w1; o = o1; break;
    case 2:  w = w2; o = o2; break;
    default: w = w3; o = o3; break;
  }
  int r0 = blockIdx.y * 32, c0 = blockIdx.x * 32;
  int tc = threadIdx.x & 31, tr = threadIdx.x >> 5;
#pragma unroll
  for (int i = 0; i < 4; ++i)
    tile[tr + i * 8][tc] = w[(size_t)(r0 + tr + i * 8) * HID_ + c0 + tc];
  __syncthreads();
#pragma unroll
  for (int i = 0; i < 4; ++i)
    o[(size_t)(c0 + tr + i * 8) * HID_ + r0 + tc] = (__bf16)tile[tc][tr + i * 8];
}

// ---------------------------------------------------------------------------
// Y = X(bf16) @ W (bf16, pre-transposed [n][k]) + b, out bf16 in [B, H, L, D].
// Double-buffered async-LDS pipeline: tile i+1 in flight behind tile i's WMMAs.
__global__ __launch_bounds__(256) void qkv_gemm_kernel(
    const __bf16* __restrict__ xq, const __bf16* __restrict__ xk,
    const __bf16* __restrict__ xv,
    const __bf16* __restrict__ Wqt, const float* __restrict__ bq,
    const __bf16* __restrict__ Wkt, const float* __restrict__ bk,
    const __bf16* __restrict__ Wvt, const float* __restrict__ bv,
    __bf16* __restrict__ qo, __bf16* __restrict__ ko, __bf16* __restrict__ vo) {
  const __bf16 *x, *Wt;
  const float* bias;
  __bf16* out;
  int z = blockIdx.z;
  if (z == 0)      { x = xq; Wt = Wqt; bias = bq; out = qo; }
  else if (z == 1) { x = xk; Wt = Wkt; bias = bk; out = ko; }
  else             { x = xv; Wt = Wvt; bias = bv; out = vo; }

  __shared__ __bf16 As[2][128 * 40];   // [m][k], padded stride 40
  __shared__ __bf16 Bs[2][128 * 40];   // [n][k], padded stride 40

  int tid = threadIdx.x;
  int wave = tid >> 5, lane = tid & 31, ln = lane & 15, hi8 = (lane >> 4) * 8;
  int m0 = blockIdx.y * 128, n0 = blockIdx.x * 128;
  int wm = (wave >> 2) * 64, wn = (wave & 3) * 32;

  auto stage_tile = [&](int k0, int buf) {
#pragma unroll
    for (int i = 0; i < 2; ++i) {      // 4 async b128 per thread per tile
      int e = tid + i * 256;
      int r = e >> 2, kk = (e & 3) * 8;
      stage16(x  + (size_t)(m0 + r) * HID_ + k0 + kk, &As[buf][r * 40 + kk]);
      stage16(Wt + (size_t)(n0 + r) * HID_ + k0 + kk, &Bs[buf][r * 40 + kk]);
    }
  };

  f32x8 acc[4][2];
#pragma unroll
  for (int mt = 0; mt < 4; ++mt)
#pragma unroll
    for (int nt = 0; nt < 2; ++nt) acc[mt][nt] = (f32x8){};

  stage_tile(0, 0);
  for (int it = 0; it < HID_ / 32; ++it) {
    int buf = it & 1;
    if (it + 1 < HID_ / 32) {
      stage_tile((it + 1) * 32, buf ^ 1);
      stage_wait_n<4>();               // tile i landed; tile i+1 stays in flight
    } else {
      stage_wait_n<0>();
    }
    __syncthreads();
    bf16x16 af[4], bfr[2];
#pragma unroll
    for (int mt = 0; mt < 4; ++mt)
      af[mt] = frag_a(&As[buf][(wm + mt * 16 + ln) * 40], hi8);
#pragma unroll
    for (int nt = 0; nt < 2; ++nt)
      bfr[nt] = frag_b(&Bs[buf][(wn + nt * 16 + ln) * 40 + hi8 * 2]);
#pragma unroll
    for (int mt = 0; mt < 4; ++mt)
#pragma unroll
      for (int nt = 0; nt < 2; ++nt)
        acc[mt][nt] = wmma_bf16(af[mt], bfr[nt], acc[mt][nt]);
    __syncthreads();
  }

#pragma unroll
  for (int nt = 0; nt < 2; ++nt) {
    int n = n0 + wn + nt * 16 + ln;
    float bn = bias[n];
    int hh = n >> 6, d = n & 63;
#pragma unroll
    for (int mt = 0; mt < 4; ++mt)
#pragma unroll
      for (int r = 0; r < 8; ++r) {
        int m = m0 + wm + mt * 16 + r + hi8;
        int bb = m >> 10, l = m & 1023;
        out[(((size_t)bb * H_ + hh) * L_ + l) * D_ + d] = (__bf16)(acc[mt][nt][r] + bn);
      }
  }
}

// ---------------------------------------------------------------------------
// One block = (b, h, 16 query rows).  8 waves x 8 tiles cover all 1024 columns.
__global__ __launch_bounds__(256) void attn_kernel(
    const __bf16* __restrict__ q, const __bf16* __restrict__ k,
    const __bf16* __restrict__ v, const __bf16* __restrict__ erel,
    float* __restrict__ probs, __bf16* __restrict__ ctx) {
  __shared__ __bf16 Pl[16 * 1024];     // normalized probs, bf16 (A-matrix for P@V)
  __shared__ __bf16 VT[64 * 136];      // transposed V chunk [d][r], padded stride
  __shared__ float  ctxs[16 * 64];
  __shared__ float  red[8][16];
  __shared__ float  rmax[16];
  __shared__ float  rsum[16];

  int tid = threadIdx.x, wave = tid >> 5, lane = tid & 31, ln = lane & 15;
  int hi = lane >> 4, hi8 = hi * 8;
  int b = blockIdx.z, h = blockIdx.y, l0 = blockIdx.x * 16;
  size_t bh = (size_t)b * H_ + h;

  const __bf16* qrow = q + (bh * L_ + l0 + ln) * D_;
  bf16x16 qa0 = frag_a(qrow, hi8);          // K = 0..31
  bf16x16 qa1 = frag_a(qrow + 32, hi8);     // K = 32..63

  f32x8 sacc[8];
#pragma unroll
  for (int t = 0; t < 8; ++t) {
    int r0 = wave * 128 + t * 16;
    const __bf16* krow = k + (bh * L_ + r0 + ln) * D_;
    f32x8 a = (f32x8){};
    a = wmma_bf16(qa0, frag_b(krow + hi * 16), a);
    a = wmma_bf16(qa1, frag_b(krow + 32 + hi * 16), a);

    // relative-position bias: G = Q_tile @ E_windowT, then lane gather
    int dbase = l0 - r0 + (L_ - 16);          // 0..2016, window rows dbase..dbase+30
    f32x8 g0 = (f32x8){}, g1 = (f32x8){};
    {
      int j0 = ln;
      const __bf16* er0 = erel + (size_t)(dbase + j0) * D_;
      g0 = wmma_bf16(qa0, frag_b(er0 + hi * 16), g0);
      g0 = wmma_bf16(qa1, frag_b(er0 + 32 + hi * 16), g0);
      int j1 = 16 + ln; if (j1 > 30) j1 = 30;   // pad column (unused) stays in range
      const __bf16* er1 = erel + (size_t)(dbase + j1) * D_;
      g1 = wmma_bf16(qa0, frag_b(er1 + hi * 16), g1);
      g1 = wmma_bf16(qa1, frag_b(er1 + 32 + hi * 16), g1);
    }
#pragma unroll
    for (int r = 0; r < 8; ++r) {
      int i = r + hi8;                 // row within tile
      int j = i - ln + 15;             // window column, 0..30
      int sl = (j & 15) | (lane & 16); // same vgpr, same half, permuted lane
      float v0 = __shfl(g0[r], sl, 32);
      float v1 = __shfl(g1[r], sl, 32);
      a[r] += (j < 16) ? v0 : v1;
    }
    sacc[t] = a;
  }

  // scale by 1/sqrt(D) and reduce row max (butterfly within 16-lane halves)
#pragma unroll
  for (int r = 0; r < 8; ++r) {
    float m = -3.0e38f;
#pragma unroll
    for (int t = 0; t < 8; ++t) {
      sacc[t][r] *= 0.125f;
      m = fmaxf(m, sacc[t][r]);
    }
#pragma unroll
    for (int s = 1; s < 16; s <<= 1) m = fmaxf(m, __shfl_xor(m, s, 32));
    if (ln == 0) red[wave][r + hi8] = m;
  }
  __syncthreads();
  if (tid < 16) {
    float m = red[0][tid];
#pragma unroll
    for (int w = 1; w < 8; ++w) m = fmaxf(m, red[w][tid]);
    rmax[tid] = m;
  }
  __syncthreads();

  // exp + row sum
#pragma unroll
  for (int r = 0; r < 8; ++r) {
    int row = r + hi8;
    float mref = rmax[row];
    float s = 0.f;
#pragma unroll
    for (int t = 0; t < 8; ++t) {
      float p = __expf(sacc[t][r] - mref);
      sacc[t][r] = p;
      s += p;
    }
#pragma unroll
    for (int sh = 1; sh < 16; sh <<= 1) s += __shfl_xor(s, sh, 32);
    if (ln == 0) red[wave][row] = s;
  }
  __syncthreads();
  if (tid < 16) {
    float s = red[0][tid];
#pragma unroll
    for (int w = 1; w < 8; ++w) s += red[w][tid];
    rsum[tid] = s;
  }
  for (int i = tid; i < 16 * 64; i += 256) ctxs[i] = 0.f;
  __syncthreads();

  // normalize; write probs (f32, single pass over HBM) and P (bf16, LDS)
#pragma unroll
  for (int r = 0; r < 8; ++r) {
    int row = r + hi8;
    int l = l0 + row;
    float inv = 1.0f / rsum[row];
    float* pout = probs + (bh * L_ + l) * (size_t)L_;
#pragma unroll
    for (int t = 0; t < 8; ++t) {
      int col = wave * 128 + t * 16 + ln;
      float p = sacc[t][r] * inv;
      pout[col] = p;
      Pl[row * 1024 + col] = (__bf16)p;
    }
  }
  __syncthreads();

  // ctx = P @ V ; 4 d-tiles x 2 k-halves across 8 waves, V chunked through LDS
  int nt = wave & 3;
  int kp = wave >> 2;
  f32x8 cacc = (f32x8){};
  for (int c = 0; c < 8; ++c) {
    const __bf16* vbase = v + (bh * L_ + c * 128) * D_;
#pragma unroll
    for (int i = 0; i < 16; ++i) {        // stage transposed V chunk
      int e = tid + i * 256;
      int rr = e >> 5, dp = e & 31;
      union { unsigned int u; __bf16 hh[2]; } cv;
      cv.u = *(const unsigned int*)(vbase + rr * D_ + dp * 2);
      VT[(dp * 2)     * 136 + rr] = cv.hh[0];
      VT[(dp * 2 + 1) * 136 + rr] = cv.hh[1];
    }
    __syncthreads();
#pragma unroll
    for (int s = 0; s < 2; ++s) {
      int kk = (kp * 2 + s) * 32;
      bf16x16 pa = frag_a(&Pl[ln * 1024 + c * 128 + kk], hi8);
      bf16x16 vbfr = frag_b(&VT[(nt * 16 + ln) * 136 + kk + hi * 16]);
      cacc = wmma_bf16(pa, vbfr, cacc);
    }
    __syncthreads();
  }
#pragma unroll
  for (int r = 0; r < 8; ++r)
    atomicAdd(&ctxs[(r + hi8) * 64 + nt * 16 + ln], cacc[r]);
  __syncthreads();
  for (int e = tid; e < 16 * 64; e += 256) {
    int row = e >> 6, d = e & 63;
    int l = l0 + row;
    ctx[(((size_t)b * L_ + l) * H_ + h) * D_ + d] = (__bf16)ctxs[e];
  }
}

// ---------------------------------------------------------------------------
// out = ctx(bf16) @ Wo (bf16 pre-transposed [n][k]) + bo, f32 output
__global__ __launch_bounds__(256) void out_gemm_kernel(
    const __bf16* __restrict__ actx, const __bf16* __restrict__ Wot,
    const float* __restrict__ bo, float* __restrict__ out) {
  __shared__ __bf16 As[2][128 * 40];
  __shared__ __bf16 Bs[2][128 * 40];

  int tid = threadIdx.x;
  int wave = tid >> 5, lane = tid & 31, ln = lane & 15, hi8 = (lane >> 4) * 8;
  int m0 = blockIdx.y * 128, n0 = blockIdx.x * 128;
  int wm = (wave >> 2) * 64, wn = (wave & 3) * 32;

  auto stage_tile = [&](int k0, int buf) {
#pragma unroll
    for (int i = 0; i < 2; ++i) {
      int e = tid + i * 256;
      int r = e >> 2, kk = (e & 3) * 8;
      stage16(actx + (size_t)(m0 + r) * HID_ + k0 + kk, &As[buf][r * 40 + kk]);
      stage16(Wot  + (size_t)(n0 + r) * HID_ + k0 + kk, &Bs[buf][r * 40 + kk]);
    }
  };

  f32x8 acc[4][2];
#pragma unroll
  for (int mt = 0; mt < 4; ++mt)
#pragma unroll
    for (int nt = 0; nt < 2; ++nt) acc[mt][nt] = (f32x8){};

  stage_tile(0, 0);
  for (int it = 0; it < HID_ / 32; ++it) {
    int buf = it & 1;
    if (it + 1 < HID_ / 32) {
      stage_tile((it + 1) * 32, buf ^ 1);
      stage_wait_n<4>();
    } else {
      stage_wait_n<0>();
    }
    __syncthreads();
    bf16x16 af[4], bfr[2];
#pragma unroll
    for (int mt = 0; mt < 4; ++mt)
      af[mt] = frag_a(&As[buf][(wm + mt * 16 + ln) * 40], hi8);
#pragma unroll
    for (int nt = 0; nt < 2; ++nt)
      bfr[nt] = frag_b(&Bs[buf][(wn + nt * 16 + ln) * 40 + hi8 * 2]);
#pragma unroll
    for (int mt = 0; mt < 4; ++mt)
#pragma unroll
      for (int nt = 0; nt < 2; ++nt)
        acc[mt][nt] = wmma_bf16(af[mt], bfr[nt], acc[mt][nt]);
    __syncthreads();
  }

#pragma unroll
  for (int nt = 0; nt < 2; ++nt) {
    int n = n0 + wn + nt * 16 + ln;
    float bn = bo[n];
#pragma unroll
    for (int mt = 0; mt < 4; ++mt)
#pragma unroll
      for (int r = 0; r < 8; ++r) {
        int m = m0 + wm + mt * 16 + r + hi8;
        out[(size_t)m * HID_ + n] = acc[mt][nt][r] + bn;
      }
  }
}

// ---------------------------------------------------------------------------
extern "C" void kernel_launch(void* const* d_in, const int* in_sizes, int n_in,
                              void* d_out, int out_size, void* d_ws, size_t ws_size,
                              hipStream_t stream) {
  (void)in_sizes; (void)n_in; (void)out_size; (void)ws_size;
  const float* xq  = (const float*)d_in[0];
  const float* xk  = (const float*)d_in[1];
  const float* xv  = (const float*)d_in[2];
  const float* Wq  = (const float*)d_in[3];
  const float* bq  = (const float*)d_in[4];
  const float* Wk  = (const float*)d_in[5];
  const float* bk  = (const float*)d_in[6];
  const float* Wv  = (const float*)d_in[7];
  const float* bv  = (const float*)d_in[8];
  const float* Wo  = (const float*)d_in[9];
  const float* bo  = (const float*)d_in[10];
  const float* rel = (const float*)d_in[11];

  char* ws = (char*)d_ws;
  const size_t MB = 1024 * 1024;
  __bf16* qb   = (__bf16*)(ws + 0 * MB);
  __bf16* kb   = (__bf16*)(ws + 8 * MB);
  __bf16* vb   = (__bf16*)(ws + 16 * MB);
  __bf16* ctxb = (__bf16*)(ws + 24 * MB);
  __bf16* xball= (__bf16*)(ws + 32 * MB);   // xq,xk,xv bf16, 8 MB each
  __bf16* wqt  = (__bf16*)(ws + 56 * MB);
  __bf16* wkt  = (__bf16*)(ws + 58 * MB);
  __bf16* wvt  = (__bf16*)(ws + 60 * MB);
  __bf16* wot  = (__bf16*)(ws + 62 * MB);
  __bf16* ereb = (__bf16*)(ws + 64 * MB);

  const size_t XE = (size_t)B_ * L_ * HID_;          // 4M elements
  __bf16* xqb = xball;
  __bf16* xkb = xball + XE;
  __bf16* xvb = xball + 2 * XE;

  float* outp  = (float*)d_out;
  float* probs = outp + XE;

  int nrel = (2 * L_ - 1) * D_;
  cvt_bf16_kernel<<<dim3((nrel + 255) / 256), dim3(256), 0, stream>>>(rel, ereb, nrel);
  cvt_x_kernel<<<dim3((unsigned)(XE / 2048), 1, 3), dim3(256), 0, stream>>>(xq, xk, xv, xball);
  cvt_wT_kernel<<<dim3(32, 32, 4), dim3(256), 0, stream>>>(Wq, Wk, Wv, Wo,
                                                           wqt, wkt, wvt, wot);
  qkv_gemm_kernel<<<dim3(HID_ / 128, (B_ * L_) / 128, 3), dim3(256), 0, stream>>>(
      xqb, xkb, xvb, wqt, bq, wkt, bk, wvt, bv, qb, kb, vb);
  attn_kernel<<<dim3(L_ / 16, H_, B_), dim3(256), 0, stream>>>(qb, kb, vb, ereb, probs, ctxb);
  out_gemm_kernel<<<dim3(HID_ / 128, (B_ * L_) / 128), dim3(256), 0, stream>>>(
      ctxb, wot, bo, outp);
}
